// GraphAttentionWithMask_21122649162386
// MI455X (gfx1250) — matrix operand order repaired and checked by reading, not compiled, not run on hardware
//
#include <hip/hip_runtime.h>
#include <hip/hip_bf16.h>
#include <math.h>

// ---------------------------------------------------------------------------
// GAT forward, fused flash-style attention, CDNA5 (gfx1250) WMMA bf16 path
// + async global->LDS tile staging (ASYNCcnt) for the attention K/V tiles.
//   N=4096, F=512, D=64, H=4, E=128, M=64
// ---------------------------------------------------------------------------

typedef __attribute__((ext_vector_type(16))) __bf16 v16bf;
typedef __attribute__((ext_vector_type(8)))  float  v8f;

#define WMMA_BF16(a, b, c) \
  __builtin_amdgcn_wmma_f32_16x16x32_bf16(false, (a), false, (b), (short)0, (c), false, false)

constexpr int NN = 4096, FF = 512, DD = 64, HH = 4, EE = 128, MM = 64;
constexpr float KNEG_INF = -9.0e15f;
constexpr float KSLOPE   = 0.2f;

// A-fragment K index for element e (16-bit A 16x32 layout, ISA 7.12.2):
// lanes 0-15: K = {0..7, 16..23}; lanes 16-31: K = {8..15, 24..31}
__device__ __forceinline__ int afrag_k(int e, int hi) {
  return hi * 8 + (e & 7) + ((e >> 3) << 4);
}

// async global -> LDS, 16 bytes per lane (GLOBAL_LOAD_ASYNC_TO_LDS_B128).
// lds_off: byte offset within the workgroup LDS allocation (VDST VGPR),
// gaddr:   64-bit global byte address (VADDR pair).
__device__ __forceinline__ void async_g2l_b128(unsigned lds_off, const void* gaddr) {
  asm volatile("global_load_async_to_lds_b128 %0, %1, off"
               :: "v"(lds_off), "v"((unsigned long long)(uintptr_t)gaddr)
               : "memory");
}
__device__ __forceinline__ void wait_async0() {
  asm volatile("s_wait_asynccnt 0" ::: "memory");
}

// ---------------------------------------------------------------------------
// K1: Wh[h] = (mask*x) @ W[h]   (4096x512 @ 512x64), per head.
// Stores f32 [h][n][d] (for f1/f2) and bf16 transposed [h][d][n] (V operand).
// grid (N/16, H), block 128 (4 waves, one 16-col d-tile per wave)
// ---------------------------------------------------------------------------
__global__ __launch_bounds__(128) void k_wh(
    const float* __restrict__ x, const float* __restrict__ mask,
    const float* __restrict__ W, float* __restrict__ whf,
    __bf16* __restrict__ whT) {
  const int h  = blockIdx.y;
  const int i0 = blockIdx.x * 16;
  const int wv = threadIdx.x >> 5;
  const int lane = threadIdx.x & 31;
  const int ln = lane & 15, hi = lane >> 4;
  const int row = i0 + ln;
  const float mv = mask[row];
  const float* Wp = W + (size_t)h * FF * DD;
  const int col = wv * 16 + ln;
  v8f c = {};
  for (int k0 = 0; k0 < FF; k0 += 32) {
    v16bf a, b;
#pragma unroll
    for (int e = 0; e < 16; ++e)
      a[e] = (__bf16)(x[(size_t)row * FF + k0 + hi * 16 + e] * mv);
    const int kb = k0 + hi * 16;
#pragma unroll
    for (int s = 0; s < 16; ++s)
      b[s] = (__bf16)Wp[(size_t)(kb + s) * DD + col];
    c = WMMA_BF16(a, b, c);
  }
#pragma unroll
  for (int r = 0; r < 8; ++r) {
    const int rr = i0 + r + 8 * hi;   // C layout: VGPR r -> row r (+8 for hi half)
    const float v = c[r];
    whf[((size_t)h * NN + rr) * DD + col] = v;
    whT[((size_t)h * DD + col) * NN + rr] = (__bf16)v;
  }
}

// ---------------------------------------------------------------------------
// K1b: f1 = Wh@a1, f2 = Wh@a2.  grid (N/256, H), block 256.
// ---------------------------------------------------------------------------
__global__ __launch_bounds__(256) void k_f12(
    const float* __restrict__ whf, const float* __restrict__ a1,
    const float* __restrict__ a2, float* __restrict__ f1,
    float* __restrict__ f2) {
  const int h = blockIdx.y;
  const int n = blockIdx.x * 256 + threadIdx.x;
  const float* r  = whf + ((size_t)h * NN + n) * DD;
  const float* A1 = a1 + h * DD;
  const float* A2 = a2 + h * DD;
  float s1 = 0.f, s2 = 0.f;
#pragma unroll 8
  for (int d = 0; d < DD; ++d) { const float v = r[d]; s1 += v * A1[d]; s2 += v * A2[d]; }
  f1[h * NN + n] = s1;
  f2[h * NN + n] = s2;
}

// ---------------------------------------------------------------------------
// K2: adjp[h][n][m] = sum_e adj[n][m][e]*aw[h][e], bf16 out.
// Single pass over the 128 MB adj tensor produces all 4 heads.
// grid (N*M/256), block 256.
// ---------------------------------------------------------------------------
__global__ __launch_bounds__(256) void k_adjp(
    const float* __restrict__ adj, const float* __restrict__ aw,
    __bf16* __restrict__ adjp) {
  __shared__ float saw[HH * EE];
  for (int i = threadIdx.x; i < HH * EE; i += 256) saw[i] = aw[i];
  __syncthreads();
  const size_t idx = (size_t)blockIdx.x * 256 + threadIdx.x;  // n*M + m
  const float* p = adj + idx * EE;
  float acc[HH] = {};
  for (int e = 0; e < EE; e += 4) {
    const float4 v = *reinterpret_cast<const float4*>(p + e);
#pragma unroll
    for (int h = 0; h < HH; ++h) {
      acc[h] += v.x * saw[h * EE + e]     + v.y * saw[h * EE + e + 1] +
                v.z * saw[h * EE + e + 2] + v.w * saw[h * EE + e + 3];
    }
  }
#pragma unroll
  for (int h = 0; h < HH; ++h) adjp[(size_t)h * NN * MM + idx] = (__bf16)acc[h];
}

// ---------------------------------------------------------------------------
// K3: fused attention per head:
//   - block (4 waves, 64 rows) cooperatively stages the shared K-tile
//     (adjp rows j0..j0+31, contiguous 4KB) and V-tile (WhT 64x32, 64B rows)
//     into LDS via GLOBAL_LOAD_ASYNC_TO_LDS_B128 (ASYNCcnt), removing the
//     4x redundant per-wave L2 traffic.
//   - S-tile = Q(16x64) @ K^T(64x32)  [WMMA, sign-only use]
//   - logits = leakyrelu(f1_i + f2_j) masked by S>0, online softmax
//   - acc   += P(16x32) @ V(32x64)    [WMMA]
// Never materializes the (N,N) matrices.  ELU + write into cat (bf16).
// grid (N/64, H), block 128.
// ---------------------------------------------------------------------------
__global__ __launch_bounds__(128) void k_attn(
    const __bf16* __restrict__ adjp, const __bf16* __restrict__ whT,
    const float* __restrict__ f1g, const float* __restrict__ f2g,
    __bf16* __restrict__ catb) {
  __shared__ __bf16 kbuf[32 * 64];      // adjp rows j0..j0+31       (4 KB)
  __shared__ __bf16 vbuf[64 * 32];      // WhT[d][j0..j0+31]         (4 KB)
  __shared__ __bf16 plds[4][16 * 32];   // per-wave P staging        (4 KB)
  const int h   = blockIdx.y;
  const int tid = threadIdx.x;
  const int wv  = tid >> 5;
  const int lane = tid & 31;
  const int ln = lane & 15, hi = lane >> 4;
  const int i0 = blockIdx.x * 64 + wv * 16;

  const unsigned kbuf_off = (unsigned)(uintptr_t)(&kbuf[0]);
  const unsigned vbuf_off = (unsigned)(uintptr_t)(&vbuf[0]);

  // Q fragments (A layout), resident for the whole j loop
  const __bf16* qp = adjp + (size_t)h * NN * MM + (size_t)(i0 + ln) * MM;
  v16bf q0, q1;
#pragma unroll
  for (int e = 0; e < 16; ++e) {
    const int k = hi * 16 + e;
    q0[e] = qp[k];
    q1[e] = qp[32 + k];
  }
  float f1r[8];
#pragma unroll
  for (int r = 0; r < 8; ++r) f1r[r] = f1g[h * NN + i0 + r + 8 * hi];

  v8f acc0 = {}, acc1 = {}, acc2 = {}, acc3 = {};
  float mrow[8], lrow[8];
#pragma unroll
  for (int r = 0; r < 8; ++r) { mrow[r] = -INFINITY; lrow[r] = 0.f; }

  const char*  kmatc = (const char*)(adjp + (size_t)h * NN * MM);
  const char*  vmatc = (const char*)(whT + (size_t)h * DD * NN);
  const float* f2p   = f2g + h * NN;
  __bf16* pw = plds[wv];

  for (int j0 = 0; j0 < NN; j0 += 32) {
    // ---- cooperative async stage of K and V tiles (each wave issues 1/4)
    {
      const char* gk = kmatc + (size_t)j0 * (MM * 2);   // contiguous 4 KB
#pragma unroll
      for (int it = 0; it < 2; ++it) {
        const int c = tid + it * 128;                   // 16B chunk id, 0..255
        async_g2l_b128(kbuf_off + c * 16, gk + (size_t)c * 16);
      }
      const char* gv = vmatc + (size_t)j0 * 2;          // 64 rows x 64 B
#pragma unroll
      for (int it = 0; it < 2; ++it) {
        const int c = tid + it * 128;                   // c = d*4 + part
        const int d = c >> 2, part = c & 3;
        async_g2l_b128(vbuf_off + d * 64 + part * 16,
                       gv + (size_t)d * (NN * 2) + part * 16);
      }
      wait_async0();
      __syncthreads();                                  // tiles published
    }
    // ---- S = Q @ K^T (two 16-col subtiles, K-dim 64 = 2 WMMAs each)
    v8f S0 = {}, S1 = {};
    {
      const __bf16* kp0 = &kbuf[(0 * 16 + ln) * MM + hi * 16];
      const __bf16* kp1 = &kbuf[(1 * 16 + ln) * MM + hi * 16];
      const v16bf b0a = *reinterpret_cast<const v16bf*>(kp0);
      const v16bf b0b = *reinterpret_cast<const v16bf*>(kp0 + 32);
      const v16bf b1a = *reinterpret_cast<const v16bf*>(kp1);
      const v16bf b1b = *reinterpret_cast<const v16bf*>(kp1 + 32);
      S0 = WMMA_BF16(q0, b0a, S0); S0 = WMMA_BF16(q1, b0b, S0);
      S1 = WMMA_BF16(q0, b1a, S1); S1 = WMMA_BF16(q1, b1b, S1);
    }
    const float f2a = f2p[j0 + ln];
    const float f2b = f2p[j0 + 16 + ln];
    // ---- mask + rank-1 logits + online softmax (per-row over 16 lanes/half)
    float p0[8], p1[8];
#pragma unroll
    for (int r = 0; r < 8; ++r) {
      float e0 = f1r[r] + f2a; e0 = e0 > 0.f ? e0 : KSLOPE * e0;
      float e1 = f1r[r] + f2b; e1 = e1 > 0.f ? e1 : KSLOPE * e1;
      const float s0 = S0[r] > 0.f ? e0 : KNEG_INF;
      const float s1 = S1[r] > 0.f ? e1 : KNEG_INF;
      float v = fmaxf(s0, s1);
      v = fmaxf(v, __shfl_xor(v, 1, 32));
      v = fmaxf(v, __shfl_xor(v, 2, 32));
      v = fmaxf(v, __shfl_xor(v, 4, 32));
      v = fmaxf(v, __shfl_xor(v, 8, 32));
      const float mnew  = fmaxf(mrow[r], v);
      const float scale = __expf(mrow[r] - mnew);
      mrow[r] = mnew;
      const float e0p = __expf(s0 - mnew);
      const float e1p = __expf(s1 - mnew);
      float rs = e0p + e1p;
      rs += __shfl_xor(rs, 1, 32);
      rs += __shfl_xor(rs, 2, 32);
      rs += __shfl_xor(rs, 4, 32);
      rs += __shfl_xor(rs, 8, 32);
      lrow[r] = lrow[r] * scale + rs;
      acc0[r] *= scale; acc1[r] *= scale; acc2[r] *= scale; acc3[r] *= scale;
      p0[r] = e0p; p1[r] = e1p;
    }
    // ---- stage P through LDS: C layout -> row-major -> A fragment
#pragma unroll
    for (int r = 0; r < 8; ++r) {
      pw[(r + 8 * hi) * 32 + ln]      = (__bf16)p0[r];
      pw[(r + 8 * hi) * 32 + 16 + ln] = (__bf16)p1[r];
    }
    __syncthreads();
    v16bf pa;
#pragma unroll
    for (int e = 0; e < 16; ++e) pa[e] = pw[ln * 32 + hi * 16 + e];
    // ---- acc += P @ V from the LDS V-tile
    {
      const __bf16* vp = &vbuf[(0 * 16 + ln) * 32 + hi * 16];
      const v16bf v0 = *reinterpret_cast<const v16bf*>(vp);
      const v16bf v1 = *reinterpret_cast<const v16bf*>(vp + 16 * 32);
      const v16bf v2 = *reinterpret_cast<const v16bf*>(vp + 32 * 32);
      const v16bf v3 = *reinterpret_cast<const v16bf*>(vp + 48 * 32);
      acc0 = WMMA_BF16(pa, v0, acc0);
      acc1 = WMMA_BF16(pa, v1, acc1);
      acc2 = WMMA_BF16(pa, v2, acc2);
      acc3 = WMMA_BF16(pa, v3, acc3);
    }
    __syncthreads();   // all waves done reading kbuf/vbuf/plds before restage
  }
  // ---- epilogue: normalize, ELU, store concatenated bf16
#pragma unroll
  for (int r = 0; r < 8; ++r) {
    const int row = i0 + r + 8 * hi;
    const float inv = 1.0f / lrow[r];
    float o0 = acc0[r] * inv, o1 = acc1[r] * inv;
    float o2 = acc2[r] * inv, o3 = acc3[r] * inv;
    o0 = o0 > 0.f ? o0 : __expf(o0) - 1.f;
    o1 = o1 > 0.f ? o1 : __expf(o1) - 1.f;
    o2 = o2 > 0.f ? o2 : __expf(o2) - 1.f;
    o3 = o3 > 0.f ? o3 : __expf(o3) - 1.f;
    __bf16* cp = catb + (size_t)row * (HH * DD) + h * DD + ln;
    cp[0] = (__bf16)o0; cp[16] = (__bf16)o1; cp[32] = (__bf16)o2; cp[48] = (__bf16)o3;
  }
}

// ---------------------------------------------------------------------------
// K4: out = leakyrelu(cat @ Wd + bd)   (4096x256 @ 256x64)
// grid (N/16), block 128 (4 waves, one 16-col d-tile each)
// ---------------------------------------------------------------------------
__global__ __launch_bounds__(128) void k_final(
    const __bf16* __restrict__ catb, const float* __restrict__ Wd,
    const float* __restrict__ bd, float* __restrict__ out) {
  const int i0 = blockIdx.x * 16;
  const int wv = threadIdx.x >> 5;
  const int lane = threadIdx.x & 31;
  const int ln = lane & 15, hi = lane >> 4;
  const int col = wv * 16 + ln;
  const __bf16* ap = catb + (size_t)(i0 + ln) * (HH * DD);
  v8f c = {};
  for (int k0 = 0; k0 < HH * DD; k0 += 32) {
    v16bf a, b;
#pragma unroll
    for (int e = 0; e < 16; ++e) a[e] = ap[k0 + hi * 16 + e];
#pragma unroll
    for (int s = 0; s < 16; ++s)
      b[s] = (__bf16)Wd[(size_t)(k0 + hi * 16 + s) * DD + col];
    c = WMMA_BF16(a, b, c);
  }
  const float bv = bd[col];
#pragma unroll
  for (int r = 0; r < 8; ++r) {
    const int row = i0 + r + 8 * hi;
    const float v = c[r] + bv;
    out[(size_t)row * DD + col] = v > 0.f ? v : KSLOPE * v;
  }
}

// ---------------------------------------------------------------------------
extern "C" void kernel_launch(void* const* d_in, const int* in_sizes, int n_in,
                              void* d_out, int out_size, void* d_ws, size_t ws_size,
                              hipStream_t stream) {
  const float* x    = (const float*)d_in[0];
  const float* mask = (const float*)d_in[1];
  const float* adj  = (const float*)d_in[2];
  const float* W    = (const float*)d_in[3];
  const float* a1   = (const float*)d_in[4];
  const float* a2   = (const float*)d_in[5];
  const float* aw   = (const float*)d_in[6];
  const float* Wd   = (const float*)d_in[7];
  const float* bd   = (const float*)d_in[8];
  float* out = (float*)d_out;

  // workspace layout (~10.1 MB; all L2-resident on MI455X's 192 MB L2)
  char* ws = (char*)d_ws;
  size_t off = 0;
  float* whf = (float*)(ws + off);   off += (size_t)HH * NN * DD * sizeof(float);   // 4 MB
  __bf16* whT = (__bf16*)(ws + off); off += (size_t)HH * NN * DD * sizeof(__bf16);  // 2 MB
  __bf16* adjp = (__bf16*)(ws + off); off += (size_t)HH * NN * MM * sizeof(__bf16); // 2 MB
  float* f1 = (float*)(ws + off);    off += (size_t)HH * NN * sizeof(float);        // 64 KB
  float* f2 = (float*)(ws + off);    off += (size_t)HH * NN * sizeof(float);        // 64 KB
  __bf16* catb = (__bf16*)(ws + off); off += (size_t)NN * HH * DD * sizeof(__bf16); // 2 MB

  k_wh   <<<dim3(NN / 16, HH), 128, 0, stream>>>(x, mask, W, whf, whT);
  k_adjp <<<dim3(NN * MM / 256), 256, 0, stream>>>(adj, aw, adjp);
  k_f12  <<<dim3(NN / 256, HH), 256, 0, stream>>>(whf, a1, a2, f1, f2);
  k_attn <<<dim3(NN / 64, HH), 128, 0, stream>>>(adjp, whT, f1, f2, catb);
  k_final<<<dim3(NN / 16), 128, 0, stream>>>(catb, Wd, bd, out);
}
